// VectorQuantizer_34651796144160
// MI455X (gfx1250) — matrix operand order repaired and checked by reading, not compiled, not run on hardware
//
#include <hip/hip_runtime.h>
#include <stdint.h>

#define K_CODES 1024
#define DDIM 256
#define NTOK 32768         // 32 * 32 * 32
#define HW 1024            // H*W
#define ROWS_PER_BLOCK 128
#define APAD 264           // 256 + 8 bf16 pad (row stride 528 B, 16B aligned)
#define EMA 0.99f
#define EPSV 1e-5f

typedef __attribute__((ext_vector_type(16))) __bf16 v16bf;
typedef __attribute__((ext_vector_type(8)))  float  v8f;

union Frag { uint4 u[2]; v16bf v; };

__device__ __forceinline__ unsigned short f2bf(float f) {
  unsigned int u = __float_as_uint(f);
  u += 0x7FFFu + ((u >> 16) & 1u);   // round-to-nearest-even
  return (unsigned short)(u >> 16);
}

// CDNA5 async global->LDS copy (tracked by ASYNCcnt)
__device__ __forceinline__ void async_ld_b128(unsigned int lds_off, const void* g) {
  asm volatile("global_load_async_to_lds_b128 %0, %1, off"
               :: "v"(lds_off), "v"((unsigned long long)(uintptr_t)g)
               : "memory");
}
__device__ __forceinline__ void wait_async0() {
  asm volatile("s_wait_asynccnt 0x0" ::: "memory");
}
__device__ __forceinline__ void wait_async8() {   // allow next batch (8 per thread) in flight
  asm volatile("s_wait_asynccnt 0x8" ::: "memory");
}

// issue one 64-code supertile (64 rows x 512 B) as 8x b128 per thread
__device__ __forceinline__ void issue_supertile(
    const unsigned short* __restrict__ ebf, int st, unsigned int ldsBase, int tid) {
  #pragma unroll
  for (int i = 0; i < 8; ++i) {
    int c2   = i * 256 + tid;               // 0..2047
    int code = c2 >> 5;
    int part = c2 & 31;
    const char* g = (const char*)ebf + (size_t)(st * 64 + code) * (DDIM * 2) + part * 16;
    async_ld_b128(ldsBase + (unsigned)(code * (APAD * 2) + part * 16), g);
  }
}

// ---------------- kernel 0: codebook -> bf16 + ||e||^2 ----------------
__global__ void __launch_bounds__(256) vq_prep_kernel(
    const float* __restrict__ emb, unsigned short* __restrict__ ebf,
    float* __restrict__ enorm)
{
  const int k = blockIdx.x;
  const int d = threadIdx.x;
  float x = emb[k * DDIM + d];
  ebf[k * DDIM + d] = f2bf(x);
  float s = x * x;
  #pragma unroll
  for (int m = 16; m >= 1; m >>= 1) s += __shfl_down(s, m, 32);
  __shared__ float red[8];
  if ((d & 31) == 0) red[d >> 5] = s;
  __syncthreads();
  if (d == 0) {
    float t = 0.f;
    #pragma unroll
    for (int i = 0; i < 8; ++i) t += red[i];
    enorm[k] = t;
  }
}

// ---------------- kernel 1: WMMA distance GEMM + argmin ----------------
__global__ void __launch_bounds__(256) vq_argmin_kernel(
    const float* __restrict__ z, const unsigned short* __restrict__ ebf,
    const float* __restrict__ enorm, int* __restrict__ idx_i32,
    float* __restrict__ idx_f32)
{
  extern __shared__ unsigned short smem[];
  unsigned short* aT  = smem;                                  // [128][APAD] z tile
  unsigned short* eT0 = smem + ROWS_PER_BLOCK * APAD;          // [64][APAD] buf 0
  unsigned short* eT1 = eT0 + 64 * APAD;                       // [64][APAD] buf 1

  const int tid  = threadIdx.x;
  const int lane = tid & 31;
  const int wv   = tid >> 5;                 // 8 waves, each owns 16 rows
  const int n0   = blockIdx.x * ROWS_PER_BLOCK;

  const unsigned int eLds0 = (unsigned int)(uintptr_t)eT0;
  const unsigned int eLds1 = (unsigned int)(uintptr_t)eT1;

  // prologue: kick off supertile 0 so HBM latency hides behind z staging
  issue_supertile(ebf, 0, eLds0, tid);

  // stage Z tile fp32 -> bf16; 4 coalesced dword loads -> one ds_store_b64
  const int rloc = tid & 127;
  {
    const int nn = n0 + rloc;
    const int b  = nn >> 10;
    const int hw = nn & (HW - 1);
    for (int j = 0; j < 32; ++j) {
      int dfeat0 = (j * 2 + (tid >> 7)) * 4;
      size_t g0 = ((size_t)(b * DDIM + dfeat0) << 10) + hw;
      float f0 = z[g0];
      float f1 = z[g0 + 1024];
      float f2 = z[g0 + 2048];
      float f3 = z[g0 + 3072];
      uint2 w;
      w.x = (unsigned)f2bf(f0) | ((unsigned)f2bf(f1) << 16);
      w.y = (unsigned)f2bf(f2) | ((unsigned)f2bf(f3) << 16);
      *(uint2*)(aT + rloc * APAD + dfeat0) = w;
    }
  }
  __syncthreads();

  // A fragments (16x32 bf16 per D-chunk) held in registers for the whole kernel.
  // layout: lanes 0-15 hold K 0-15; lanes 16-31 hold K 16-31 (matches B fill).
  Frag aF[8];
  const unsigned short* aRow = aT + (wv * 16 + (lane & 15)) * APAD;
  const int khalf = (lane >> 4) * 16;
  #pragma unroll
  for (int c = 0; c < 8; ++c) {
    aF[c].u[0] = *(const uint4*)(aRow + c * 32 + khalf);
    aF[c].u[1] = *(const uint4*)(aRow + c * 32 + 8 + khalf);
  }

  float rmin[8];
  int   ridx[8];
  #pragma unroll
  for (int r = 0; r < 8; ++r) { rmin[r] = 3.4e38f; ridx[r] = 0x7fffffff; }

  for (int st = 0; st < 16; ++st) {
    const unsigned short* eTp = (st & 1) ? eT1 : eT0;

    // separate compute(st-1) readers of buf (1-p) from the writes below
    __syncthreads();

    // preload e-norms for this supertile (latency hidden behind WMMA chain)
    float en[4];
    #pragma unroll
    for (int kt = 0; kt < 4; ++kt)
      en[kt] = enorm[st * 64 + kt * 16 + (lane & 15)];

    if (st + 1 < 16) {
      issue_supertile(ebf, st + 1, ((st + 1) & 1) ? eLds1 : eLds0, tid);
      wait_async8();               // batch st done; batch st+1 stays in flight
    } else {
      wait_async0();
    }
    __syncthreads();               // all waves' batch-st data visible in LDS

    for (int kt = 0; kt < 4; ++kt) {
      const int cbase = st * 64 + kt * 16;
      // B 32x16 bf16: lane n = code column (n&15); lanes 0-15 hold K0-15, 16-31 hold K16-31
      const unsigned short* eRow = eTp + (kt * 16 + (lane & 15)) * APAD + (lane >> 4) * 16;

      v8f acc = {0.f, 0.f, 0.f, 0.f, 0.f, 0.f, 0.f, 0.f};
      Frag bF[2];
      bF[0].u[0] = *(const uint4*)(eRow + 0);
      bF[0].u[1] = *(const uint4*)(eRow + 8);
      #pragma unroll
      for (int c = 0; c < 8; ++c) {           // depth-2 pipelined B loads
        if (c < 7) {
          bF[(c + 1) & 1].u[0] = *(const uint4*)(eRow + (c + 1) * 32);
          bF[(c + 1) & 1].u[1] = *(const uint4*)(eRow + (c + 1) * 32 + 8);
        }
        acc = __builtin_amdgcn_wmma_f32_16x16x32_bf16(
                false, aF[c].v, false, bF[c & 1].v, (short)0, acc, false, false);
      }

      const int code = cbase + (lane & 15);
      #pragma unroll
      for (int r = 0; r < 8; ++r) {
        float d = en[kt] - 2.0f * acc[r];     // ||z||^2 dropped (constant per row)
        if (d < rmin[r]) { rmin[r] = d; ridx[r] = code; }  // strict < keeps first index
      }
    }
  }

  // reduce over the 16-lane N groups (xor 1..8 stays within each half)
  #pragma unroll
  for (int r = 0; r < 8; ++r) {
    float v = rmin[r]; int ix = ridx[r];
    #pragma unroll
    for (int m = 1; m <= 8; m <<= 1) {
      float ov = __shfl_xor(v, m, 32);
      int   oi = __shfl_xor(ix, m, 32);
      if (ov < v || (ov == v && oi < ix)) { v = ov; ix = oi; }
    }
    if ((lane & 15) == 0) {
      int n = n0 + wv * 16 + r + ((lane >> 4) << 3);  // VGPR r -> M=r / M=r+8
      idx_i32[n] = ix;
      idx_f32[n] = (float)ix;
    }
  }
}

// ------------- kernel 2: z_q gather, loss, counts, embed_sum -------------
__global__ void __launch_bounds__(256) vq_scatter_kernel(
    const float* __restrict__ z, const float* __restrict__ emb,
    const int* __restrict__ idx, float* __restrict__ zq_out,
    float* __restrict__ counts, float* __restrict__ esum,
    float* __restrict__ loss_acc)
{
  const int tid  = threadIdx.x;
  const int nl   = tid & 127;
  const int half = tid >> 7;
  const int n  = blockIdx.x * 128 + nl;
  const int k  = idx[n];
  const int b  = n >> 10;
  const int hw = n & (HW - 1);
  if (half == 0) atomicAdd(&counts[k], 1.0f);
  float lsum = 0.f;
  for (int d = half; d < DDIM; d += 2) {
    float e  = emb[k * DDIM + d];
    size_t g = ((size_t)(b * DDIM + d) << 10) + hw;
    float zv = z[g];
    __builtin_nontemporal_store(zv + (e - zv), &zq_out[g]);  // ST value, write-once
    float df = zv - e;
    lsum += df * df;
    atomicAdd(&esum[k * DDIM + d], zv);
  }
  #pragma unroll
  for (int m = 16; m >= 1; m >>= 1) lsum += __shfl_down(lsum, m, 32);
  __shared__ float red[8];
  if ((tid & 31) == 0) red[tid >> 5] = lsum;
  __syncthreads();
  if (tid == 0) {
    float t = 0.f;
    #pragma unroll
    for (int i = 0; i < 8; ++i) t += red[i];
    atomicAdd(loss_acc, t);
  }
}

// ------------- kernel 3a: new_cs, n = sum(new_cs), final loss -------------
__global__ void __launch_bounds__(1024) vq_ema_cs_kernel(
    const float* __restrict__ cs, const float* __restrict__ counts,
    const float* __restrict__ loss_acc,
    float* __restrict__ new_cs, float* __restrict__ loss_out,
    float* __restrict__ n_out)
{
  const int k = threadIdx.x;
  float v = EMA * cs[k] + (1.0f - EMA) * counts[k];
  new_cs[k] = v;
  float s = v;
  #pragma unroll
  for (int m = 16; m >= 1; m >>= 1) s += __shfl_down(s, m, 32);
  __shared__ float red[32];
  if ((k & 31) == 0) red[k >> 5] = s;
  __syncthreads();
  if (k == 0) {
    float t = 0.f;
    #pragma unroll
    for (int i = 0; i < 32; ++i) t += red[i];
    *n_out = t;
    *loss_out = *loss_acc * (1.0f / 8388608.0f);   // mean over B*D*H*W
  }
}

// ------------- kernel 3b: new_es + new_embedding -------------
__global__ void __launch_bounds__(256) vq_ema_emb_kernel(
    const float* __restrict__ ema_es, const float* __restrict__ esum,
    const float* __restrict__ new_cs, const float* __restrict__ n_in,
    float* __restrict__ new_es, float* __restrict__ new_emb)
{
  const int k = blockIdx.x;
  const int d = threadIdx.x;
  float nval = *n_in;
  float csv  = new_cs[k];
  float cluster = (csv + EPSV) / (nval + (float)K_CODES * EPSV) * nval;
  float es = EMA * ema_es[k * DDIM + d] + (1.0f - EMA) * esum[k * DDIM + d];
  new_es[k * DDIM + d]  = es;
  new_emb[k * DDIM + d] = es / cluster;
}

extern "C" void kernel_launch(void* const* d_in, const int* in_sizes, int n_in,
                              void* d_out, int out_size, void* d_ws, size_t ws_size,
                              hipStream_t stream) {
  const float* z   = (const float*)d_in[0];
  const float* emb = (const float*)d_in[1];
  const float* cs  = (const float*)d_in[2];
  const float* es  = (const float*)d_in[3];

  char* ws = (char*)d_ws;
  unsigned short* ebf = (unsigned short*)(ws + 0);        // 524288 B  (1024x256 bf16)
  float* enorm  = (float*)(ws + 524288);                  // 4096  B
  int*   idx    = (int*)  (ws + 528384);                  // 131072 B
  float* counts = (float*)(ws + 659456);                  // 4096  B  (zeroed)
  float* esum   = (float*)(ws + 663552);                  // 1048576 B (zeroed)
  float* lossa  = (float*)(ws + 1712128);                 // 4 B (zeroed)
  float* nval   = (float*)(ws + 1712132);                 // 4 B (zeroed)

  float* out    = (float*)d_out;
  float* o_zq   = out;               // 8388608 : z_q_st (B,D,H,W)
  float* o_loss = out + 8388608;     // 1       : commitment loss
  float* o_idx  = out + 8388609;     // 32768   : indices (as float values)
  float* o_nemb = out + 8421377;     // 262144  : new_embedding
  float* o_ncs  = out + 8683521;     // 1024    : new_cs
  float* o_nes  = out + 8684545;     // 262144  : new_es

  // zero counts / embed_sum / loss / n accumulators (contiguous region)
  hipMemsetAsync(counts, 0, 4096 + 1048576 + 8, stream);

  vq_prep_kernel<<<K_CODES, 256, 0, stream>>>(emb, ebf, enorm);

  const size_t lds_bytes = (size_t)(ROWS_PER_BLOCK + 128) * APAD * sizeof(unsigned short); // 135168
  vq_argmin_kernel<<<NTOK / ROWS_PER_BLOCK, 256, lds_bytes, stream>>>(
      z, ebf, enorm, idx, o_idx);

  vq_scatter_kernel<<<NTOK / 128, 256, 0, stream>>>(z, emb, idx, o_zq, counts, esum, lossa);
  vq_ema_cs_kernel<<<1, 1024, 0, stream>>>(cs, counts, lossa, o_ncs, o_loss, nval);
  vq_ema_emb_kernel<<<K_CODES, 256, 0, stream>>>(es, esum, o_ncs, nval, o_nes, o_nemb);
}
